// GQAttention_27470610825611
// MI455X (gfx1250) — matrix operand order repaired and checked by reading, not compiled, not run on hardware
//
#include <hip/hip_runtime.h>
#include <hip/hip_bf16.h>

// ---------------------------------------------------------------------------
// GQA attention, CDNA5 (gfx1250, wave32) WMMA implementation.
//
// B=128 T=384 C=256 H=4 KV=2 D=64.  All matmuls use v_wmma_f32_16x16x32_f16.
// Roofline: ~38.6 GFLOP total, working set ~100 MB f16 (fits 192 MB L2);
// flash-style online softmax avoids ~600 MB of score traffic.  Everything the
// WMMA B-operand needs is pre-transposed in global memory (weights at cvt
// time, V in the projection epilogue), so B-fragments are single contiguous
// 32-byte loads and no LDS staging / block barriers are needed anywhere
// except the per-wave P tile (C-layout -> A-layout transpose).
// ---------------------------------------------------------------------------

#define DEVINL __device__ __forceinline__

typedef _Float16 half_t;
typedef __attribute__((ext_vector_type(16))) _Float16 v16h;
typedef __attribute__((ext_vector_type(8)))  _Float16 v8h;
typedef __attribute__((ext_vector_type(8)))  float    v8f;

static constexpr int Bb = 128, Tt = 384, Cc = 256, Hh = 4, KVv = 2, Dd = 64;

DEVINL v8f wmma_f16(v16h a, v16h b, v8f c) {
  // (neg_a, A, neg_b, B, c_mod, C, reuse_a, reuse_b)
  return __builtin_amdgcn_wmma_f32_16x16x32_f16(false, a, false, b, (short)0, c,
                                                false, false);
}

DEVINL v16h cat8(v8h lo, v8h hi) {
  return __builtin_shufflevector(lo, hi, 0, 1, 2, 3, 4, 5, 6, 7, 8, 9, 10, 11,
                                 12, 13, 14, 15);
}

// A-fragment (16x32 f16) from a row-major f16 row pointer for this lane.
// lane<16: K = {kb..kb+15}; lane>=16: K = {kb+16..kb+31}.
DEVINL v16h load_a32(const half_t* rowptr, int kb, int kh) {
  const half_t* p = rowptr + kb + kh * 16;
  v8h lo = *(const v8h*)p;
  v8h hi = *(const v8h*)(p + 8);
  return cat8(lo, hi);
}

// ---------------------------------------------------------------------------
// f32 -> f16 conversion, 8-wide vectorized (b128 in, b128 out, packed cvt).
// n must be a multiple of 8 (true for all buffers here).
// ---------------------------------------------------------------------------
__global__ void cvt_f16_kernel(const float* __restrict__ src,
                               half_t* __restrict__ dst, int n8) {
  int i = blockIdx.x * blockDim.x + threadIdx.x;
  if (i < n8) {
    v8f f = *(const v8f*)(src + (size_t)i * 8);
    v8h h;
#pragma unroll
    for (int e = 0; e < 8; ++e) h[e] = (half_t)f[e];
    *(v8h*)(dst + (size_t)i * 8) = h;
  }
}

// dst[n][k] = (f16) src[k][n];  K == 256 rows, N columns (weights are tiny,
// permanently cache-resident; scalar transpose is fine).
__global__ void cvt_t_f16_kernel(const float* __restrict__ src,
                                 half_t* __restrict__ dst, int N) {
  int i = blockIdx.x * blockDim.x + threadIdx.x;
  if (i < Cc * N) {
    int k = i / N, n = i - k * N;
    dst[n * Cc + k] = (half_t)src[i];
  }
}

// ---------------------------------------------------------------------------
// Projection GEMM: OUT16 = X16[49152x256] @ W, with W given transposed
// (WT[n][k]) so B-fragments are contiguous 32B loads.  Fused RoPE epilogue.
// MODE 0: Q (rope, [B,H,T,D]);  1: K (rope, [B,KV,T,D]);
// MODE 2: V (no rope, written TRANSPOSED as [B,KV,D,T] for the PV matmul).
// Block = 8 waves, 128 rows x 64 cols; blockIdx.y selects the head (tile==D).
// ---------------------------------------------------------------------------
template <int MODE>
__global__ __launch_bounds__(256) void proj_kernel(
    const half_t* __restrict__ X, const half_t* __restrict__ WT,
    half_t* __restrict__ OUT) {
  const int tid = threadIdx.x;
  const int wave = tid >> 5, lane = tid & 31;
  const int kh = lane >> 4, ln = lane & 15;
  const int n0 = blockIdx.y * 64;
  const int mbase = blockIdx.x * 128 + wave * 16;
  const half_t* xrow = X + (mbase + ln) * Cc;

  v8f acc[4] = {};
#pragma unroll
  for (int kc = 0; kc < 8; ++kc) {
    const int kb = kc * 32;
    v16h a = load_a32(xrow, kb, kh);
#pragma unroll
    for (int nt = 0; nt < 4; ++nt) {
      v16h b = *(const v16h*)(WT + (n0 + nt * 16 + ln) * Cc + kb + kh * 16);
      acc[nt] = wmma_f16(a, b, acc[nt]);
    }
  }

  // Epilogue: C-layout row = j + 8*kh, col = nt*16 + ln.  RoPE pairs (d,d^32)
  // live in tiles nt and nt^2 of the same lane.
  const int heads = (MODE == 0) ? Hh : KVv;
#pragma unroll
  for (int j = 0; j < 8; ++j) {
    const int m = mbase + j + 8 * kh;
    const int bb = m / Tt, t = m % Tt;
#pragma unroll
    for (int nt = 0; nt < 4; ++nt) {
      const int d = nt * 16 + ln;
      float v = acc[nt][j];
      if (MODE == 2) {
        // V: transposed layout [B,KV,D,T]
        OUT[((bb * KVv + (int)blockIdx.y) * Dd + d) * Tt + t] = (half_t)v;
      } else {
        float vp = acc[nt ^ 2][j];
        // inv_freq = 10000^(-(d%32)/32)
        float inv = __expf(-(float)(d & 31) * 0.03125f * 9.210340371976184f);
        float sn, cs;
        __sincosf((float)t * inv, &sn, &cs);
        float r = (d < 32) ? (v * cs - vp * sn) : (v * cs + vp * sn);
        OUT[((bb * heads + (int)blockIdx.y) * Tt + t) * Dd + d] = (half_t)r;
      }
    }
  }
}

// ---------------------------------------------------------------------------
// Flash attention.  Block = one (b, h, 128-query-row slab); 8 independent
// waves, each owning 16 query rows.  Keys/values streamed in 64-wide chunks
// straight from global (L2-resident); online softmax with per-row running
// max/sum.  Only LDS use: per-wave P tile (C-layout -> A-layout transpose),
// which is same-wave in-order, so no barriers at all.
// ---------------------------------------------------------------------------
__global__ __launch_bounds__(256) void attn_kernel(
    const half_t* __restrict__ Q, const half_t* __restrict__ K,
    const half_t* __restrict__ VT, half_t* __restrict__ Y) {
  __shared__ half_t P[8][16][72];  // per-wave P tile, 16B-aligned rows

  const int tid = threadIdx.x, wave = tid >> 5, lane = tid & 31;
  const int kh = lane >> 4, ln = lane & 15;
  const int r = (int)blockIdx.x % 3;
  const int bh = (int)blockIdx.x / 3;
  const int h = bh % Hh, b = bh / Hh;
  const int kv = h >> 1;  // REP = 2
  const int q0 = r * 128 + wave * 16;

  const half_t* Qb = Q + ((b * Hh + h) * Tt) * Dd;
  const half_t* Kb = K + ((b * KVv + kv) * Tt) * Dd;   // [T][D]
  const half_t* Vb = VT + ((b * KVv + kv) * Dd) * Tt;  // [D][T]

  // Q fragments for both 32-wide K-dim chunks of D=64 (held for whole kernel).
  v16h aq[2];
  {
    const half_t* qrow = Qb + (q0 + ln) * Dd;
    aq[0] = load_a32(qrow, 0, kh);
    aq[1] = load_a32(qrow, 32, kh);
  }

  float m_r[8], l_r[8];
  v8f accy[4] = {};
#pragma unroll
  for (int j = 0; j < 8; ++j) { m_r[j] = -1e30f; l_r[j] = 0.0f; }

  const int nchunks = (r + 1) * 2;  // 64-key chunks needed under causal mask
  for (int ch = 0; ch < nchunks; ++ch) {
    const int s0 = ch * 64;

    // Prefetch next chunk of K (rows) and V (transposed rows) into caches.
    if (ch + 1 < nchunks) {
      const int sn = s0 + 64;
      __builtin_prefetch(Kb + (sn + lane * 2) * Dd, 0, 0);
      __builtin_prefetch(Vb + (lane * 2) * Tt + sn, 0, 0);
    }

    // S = Q @ K^T for this 16x64 chunk.  K stored [T,D] => B-fragment is one
    // contiguous 32B load per lane (column = key, K-dim = d).
    v8f sa[4] = {};
#pragma unroll
    for (int st = 0; st < 4; ++st) {
      const half_t* krow = Kb + (s0 + st * 16 + ln) * Dd;
#pragma unroll
      for (int c = 0; c < 2; ++c) {
        v16h bk = *(const v16h*)(krow + c * 32 + kh * 16);
        sa[st] = wmma_f16(aq[c], bk, sa[st]);
      }
    }

    // Scale + causal mask + chunk row-max (rows spread over 16 lanes).
    float cmax[8];
#pragma unroll
    for (int j = 0; j < 8; ++j) cmax[j] = -1e30f;
#pragma unroll
    for (int st = 0; st < 4; ++st) {
      const int s = s0 + st * 16 + ln;
#pragma unroll
      for (int j = 0; j < 8; ++j) {
        const int t = q0 + j + 8 * kh;
        float v = sa[st][j] * 0.125f;  // 1/sqrt(64)
        v = (s <= t) ? v : -1e30f;
        sa[st][j] = v;
        cmax[j] = fmaxf(cmax[j], v);
      }
    }
#pragma unroll
    for (int j = 0; j < 8; ++j)
#pragma unroll
      for (int off = 1; off < 16; off <<= 1)
        cmax[j] = fmaxf(cmax[j], __shfl_xor(cmax[j], off, 32));

    // Online-softmax update.
    float alpha[8];
#pragma unroll
    for (int j = 0; j < 8; ++j) {
      float mn = fmaxf(m_r[j], cmax[j]);
      alpha[j] = __expf(m_r[j] - mn);
      m_r[j] = mn;
    }
#pragma unroll
    for (int nt = 0; nt < 4; ++nt)
#pragma unroll
      for (int j = 0; j < 8; ++j) accy[nt][j] *= alpha[j];

    float rsum[8];
#pragma unroll
    for (int j = 0; j < 8; ++j) rsum[j] = 0.0f;
#pragma unroll
    for (int st = 0; st < 4; ++st)
#pragma unroll
      for (int j = 0; j < 8; ++j) {
        float p = __expf(sa[st][j] - m_r[j]);
        sa[st][j] = p;
        rsum[j] += p;
      }
#pragma unroll
    for (int j = 0; j < 8; ++j) {
#pragma unroll
      for (int off = 1; off < 16; off <<= 1)
        rsum[j] += __shfl_xor(rsum[j], off, 32);
      l_r[j] = l_r[j] * alpha[j] + rsum[j];
    }

    // P (C-layout) -> LDS (row-major f16).  Same wave writes & reads; LDS ops
    // are in-order per wave, no barrier needed.
#pragma unroll
    for (int st = 0; st < 4; ++st)
#pragma unroll
      for (int j = 0; j < 8; ++j)
        P[wave][j + 8 * kh][st * 16 + ln] = (half_t)sa[st][j];

    // Y += P[16x64] @ V[64x64]:  A from P-LDS, B straight from transposed V
    // in global ([D][T]: K-dim s contiguous at fixed d => 32B load per lane).
#pragma unroll
    for (int c = 0; c < 2; ++c) {
      v16h ap = load_a32(&P[wave][ln][0], c * 32, kh);
#pragma unroll
      for (int nt = 0; nt < 4; ++nt) {
        v16h bv =
            *(const v16h*)(Vb + (nt * 16 + ln) * Tt + s0 + c * 32 + kh * 16);
        accy[nt] = wmma_f16(ap, bv, accy[nt]);
      }
    }
  }

  // Normalize and write Y as f16 in [B,T,C] order for the output projection.
#pragma unroll
  for (int j = 0; j < 8; ++j) {
    const float inv = 1.0f / l_r[j];
    const int t = q0 + j + 8 * kh;
    half_t* yrow = Y + (b * Tt + t) * Cc + h * Dd;
#pragma unroll
    for (int nt = 0; nt < 4; ++nt)
      yrow[nt * 16 + ln] = (half_t)(accy[nt][j] * inv);
  }
}

// ---------------------------------------------------------------------------
// Output projection: OUT(f32)[49152x256] = Y16 @ Wo, Wo given transposed.
// ---------------------------------------------------------------------------
__global__ __launch_bounds__(256) void outproj_kernel(
    const half_t* __restrict__ Y, const half_t* __restrict__ WT,
    float* __restrict__ OUT) {
  const int tid = threadIdx.x;
  const int wave = tid >> 5, lane = tid & 31;
  const int kh = lane >> 4, ln = lane & 15;
  const int n0 = blockIdx.y * 64;
  const int mbase = blockIdx.x * 128 + wave * 16;
  const half_t* yrow = Y + (mbase + ln) * Cc;

  v8f acc[4] = {};
#pragma unroll
  for (int kc = 0; kc < 8; ++kc) {
    const int kb = kc * 32;
    v16h a = load_a32(yrow, kb, kh);
#pragma unroll
    for (int nt = 0; nt < 4; ++nt) {
      v16h b = *(const v16h*)(WT + (n0 + nt * 16 + ln) * Cc + kb + kh * 16);
      acc[nt] = wmma_f16(a, b, acc[nt]);
    }
  }

#pragma unroll
  for (int j = 0; j < 8; ++j) {
    const int m = mbase + j + 8 * kh;
    float* orow = OUT + (size_t)m * Cc + n0;
#pragma unroll
    for (int nt = 0; nt < 4; ++nt) orow[nt * 16 + ln] = acc[nt][j];
  }
}

// ---------------------------------------------------------------------------
// Host side.  Inputs: x[B,T,C] f32, Wq[C,256], Wk[C,128], Wv[C,128], Wo[C,C].
// Workspace layout (f16 halves), ~102 MB total.
// ---------------------------------------------------------------------------
extern "C" void kernel_launch(void* const* d_in, const int* in_sizes, int n_in,
                              void* d_out, int out_size, void* d_ws,
                              size_t ws_size, hipStream_t stream) {
  (void)in_sizes; (void)n_in; (void)out_size; (void)ws_size;
  const float* x  = (const float*)d_in[0];
  const float* Wq = (const float*)d_in[1];
  const float* Wk = (const float*)d_in[2];
  const float* Wv = (const float*)d_in[3];
  const float* Wo = (const float*)d_in[4];
  float* out = (float*)d_out;

  const int nX  = Bb * Tt * Cc;        // 12,582,912
  const int nWq = Cc * Hh * Dd;        // 65,536
  const int nWk = Cc * KVv * Dd;       // 32,768
  const int nQ  = Bb * Hh * Tt * Dd;   // 12,582,912
  const int nK  = Bb * KVv * Tt * Dd;  // 6,291,456
  const int nWo = Cc * Cc;             // 65,536

  half_t* p = (half_t*)d_ws;
  half_t* x16   = p;  p += nX;
  half_t* WqT16 = p;  p += nWq;   // transposed: [N=256][K=256]
  half_t* WkT16 = p;  p += nWk;   // transposed: [N=128][K=256]
  half_t* WvT16 = p;  p += nWk;
  half_t* WoT16 = p;  p += nWo;
  half_t* Q16   = p;  p += nQ;    // [B,H,T,D]
  half_t* K16   = p;  p += nK;    // [B,KV,T,D]
  half_t* VT16  = p;  p += nK;    // [B,KV,D,T]  (transposed for PV B-frags)
  half_t* Y16   = p;  p += nX;    // [B,T,C]

  // 1) f32 -> f16 conversions; x is converted 8-wide (b128 loads/stores),
  //    weights are transpose-converted so WMMA B-fragments read contiguously.
  cvt_f16_kernel<<<(nX / 8 + 255) / 256, 256, 0, stream>>>(x, x16, nX / 8);
  cvt_t_f16_kernel<<<(nWq + 255) / 256, 256, 0, stream>>>(Wq, WqT16, Hh * Dd);
  cvt_t_f16_kernel<<<(nWk + 255) / 256, 256, 0, stream>>>(Wk, WkT16, KVv * Dd);
  cvt_t_f16_kernel<<<(nWk + 255) / 256, 256, 0, stream>>>(Wv, WvT16, KVv * Dd);
  cvt_t_f16_kernel<<<(nWo + 255) / 256, 256, 0, stream>>>(Wo, WoT16, Cc);

  // 2) QKV projections with fused RoPE (M = 49152 rows / 128 per block).
  proj_kernel<0><<<dim3(384, 4), 256, 0, stream>>>(x16, WqT16, Q16);
  proj_kernel<1><<<dim3(384, 2), 256, 0, stream>>>(x16, WkT16, K16);
  proj_kernel<2><<<dim3(384, 2), 256, 0, stream>>>(x16, WvT16, VT16);

  // 3) Flash attention: B*H*(T/128) = 1536 blocks.
  attn_kernel<<<Bb * Hh * 3, 256, 0, stream>>>(Q16, K16, VT16, Y16);

  // 4) Output projection -> f32 d_out.
  outproj_kernel<<<dim3(384, 4), 256, 0, stream>>>(Y16, WoT16, out);
}
